// CharCTCDecoder_76544907149497
// MI455X (gfx1250) — matrix operand order repaired and checked by reading, not compiled, not run on hardware
//
#include <hip/hip_runtime.h>
#include <hip/hip_bf16.h>
#include <math.h>

// ---------------------------------------------------------------------------
// MI455X (gfx1250) CharCTCDecoder — bf16x3 WMMA GEMMs with async-to-LDS staging
//   prep: W_exp/W1/W2 -> transposed [N][K] bf16 hi/lo planes (once)
//         rep         -> flat bf16 hi/lo planes (once)
//   G1: rep @ W_exp + b_exp  -> states as chunk-blocked hi/lo planes in d_out
//   G2: states @ W1 + b1,relu-> h hi/lo planes in ws (L2-resident chunk)
//   G3: h @ W2 + b2          -> fp32 logits in d_out (overwrites consumed states)
//   log_softmax rows, CTC forward DP, mean.
// GEMM inner loop: 8x global_load_async_to_lds_b128 (ASYNCcnt, double-buffered)
// + 24x ds_load_b128 + 24x v_wmma_f32_16x16x32_bf16 per 128x128x32 step.
// All dims are multiples of the tile sizes -> no bounds checks; EXEC all-ones.
// ---------------------------------------------------------------------------

typedef __attribute__((ext_vector_type(16))) __bf16 bf16x16;
typedef __attribute__((ext_vector_type(8)))  float  v8f;

struct alignas(16) U128 { unsigned int x, y, z, w; };
union  FragU { U128 q[2]; bf16x16 v; };

#define NEGV (-1e30f)

// round-to-nearest-even bf16 split: x ~= hi + lo
__device__ __forceinline__ void bf16_split(float x, unsigned short& hi, unsigned short& lo) {
  unsigned u = __float_as_uint(x);
  unsigned r = u + 0x7FFFu + ((u >> 16) & 1u);
  unsigned short h = (unsigned short)(r >> 16);
  float hf = __uint_as_float(((unsigned)h) << 16);
  float rem = x - hf;
  unsigned u2 = __float_as_uint(rem);
  unsigned r2 = u2 + 0x7FFFu + ((u2 >> 16) & 1u);
  hi = h;
  lo = (unsigned short)(r2 >> 16);
}

// elementwise fp32 -> (hi,lo) planes
__global__ void convert_plane_kernel(const float* __restrict__ src,
                                     unsigned short* __restrict__ hi,
                                     unsigned short* __restrict__ lo, int n) {
  int i = blockIdx.x * blockDim.x + threadIdx.x;
  if (i < n) {
    unsigned short h, l;
    bf16_split(src[i], h, l);
    hi[i] = h; lo[i] = l;
  }
}

// W[K,N] fp32 -> transposed planes hiT/loT[N,K]
__global__ void convert_transpose_kernel(const float* __restrict__ W,
                                         unsigned short* __restrict__ hiT,
                                         unsigned short* __restrict__ loT, int K, int N) {
  int i = blockIdx.x * blockDim.x + threadIdx.x;  // i = n*K + k
  if (i < K * N) {
    int n = i / K, k = i - n * K;
    unsigned short h, l;
    bf16_split(W[(size_t)k * N + n], h, l);
    hiT[i] = h; loT[i] = l;
  }
}

// C[M,N] = A[M,K] @ W[K,N] + bias, A given as bf16 hi/lo planes [M,K],
// W as transposed bf16 hi/lo planes [N,K].
// outMode: 0 = fp32 C (stride N); 1 = packed hi/lo planes (stride N);
//          2 = chunk-blocked packed planes (G1 states layout).
// Grid: (N/128, rows/128), 256 threads = 8 waves (2x4), wave computes 64x32.
__global__ __launch_bounds__(256)
void gemm_wmma_async(const unsigned short* __restrict__ aHiP, const unsigned short* __restrict__ aLoP,
                     const unsigned short* __restrict__ bHiP, const unsigned short* __restrict__ bLoP,
                     const float* __restrict__ bias,
                     float* __restrict__ Cf,
                     unsigned short* __restrict__ cHi, unsigned short* __restrict__ cLo,
                     int N, int K, int relu, int outMode, int Dd, int Vv, int CHc) {
  // [buf][plane: aHi,aLo,bHi,bLo][128 rows x 40 halves] (rows padded, 16B aligned)
  __shared__ alignas(16) unsigned short smem[2][4][128 * 40];

  const int tid     = threadIdx.x;
  const int lane    = tid & 31;
  const int wave    = tid >> 5;
  const int waveRow = wave >> 2;   // 0..1
  const int waveCol = wave & 3;    // 0..3
  const int mrow    = lane & 15;   // A: M / B,C: N
  const int half    = lane >> 4;
  const int rowBase = blockIdx.y * 128;
  const int colBase = blockIdx.x * 128;

  // per-thread async copy descriptors: 4 A-units + 4 B-units of 16B each per chunk
  // unit u = q*256+tid ; plane p=u>>9 ; rem=u&511 ; r=rem>>2 ; part=rem&3
  unsigned ldsA[4], ldsB[4];
  unsigned long long gA[4], gB[4];
  const unsigned smemBase = (unsigned)(unsigned long long)(uintptr_t)&smem[0][0][0];
#pragma unroll
  for (int q = 0; q < 4; ++q) {
    unsigned u = (unsigned)(q * 256 + tid);
    unsigned p = u >> 9, rem = u & 511u, r = rem >> 2, part = rem & 3u;
    ldsA[q] = smemBase + 2u * (p * 5120u + r * 40u + part * 8u);
    const unsigned short* ap = p ? aLoP : aHiP;
    gA[q] = (unsigned long long)(uintptr_t)(ap + (size_t)(rowBase + r) * K + part * 8);
    ldsB[q] = smemBase + 2u * ((2u + p) * 5120u + r * 40u + part * 8u);
    const unsigned short* bp = p ? bLoP : bHiP;
    gB[q] = (unsigned long long)(uintptr_t)(bp + (size_t)(colBase + r) * K + part * 8);
  }

  auto issue = [&](int buf, int kt) {
    unsigned bo = (unsigned)buf * 40960u;            // LDS bytes per buffer
    unsigned long long ko = (unsigned long long)kt * 2u;  // global byte advance along K
#pragma unroll
    for (int q = 0; q < 4; ++q) {
      asm volatile("global_load_async_to_lds_b128 %0, %1, off"
                   :: "v"(ldsA[q] + bo), "v"(gA[q] + ko) : "memory");
      asm volatile("global_load_async_to_lds_b128 %0, %1, off"
                   :: "v"(ldsB[q] + bo), "v"(gB[q] + ko) : "memory");
    }
  };

  v8f acc[4][2];
#pragma unroll
  for (int i = 0; i < 4; ++i)
#pragma unroll
    for (int j = 0; j < 2; ++j)
#pragma unroll
      for (int v = 0; v < 8; ++v) acc[i][j][v] = 0.f;

  issue(0, 0);
  int buf = 0;
  for (int kt = 0; kt < K; kt += 32) {
    asm volatile("s_wait_asynccnt 0x0" ::: "memory");  // my fills for smem[buf] done
    __syncthreads();                                   // everyone's fills visible
    if (kt + 32 < K) issue(buf ^ 1, kt + 32);          // overlap next fill with WMMAs

    const unsigned short* aHiL = &smem[buf][0][0];
    const unsigned short* aLoL = &smem[buf][1][0];
    const unsigned short* bHiL = &smem[buf][2][0];
    const unsigned short* bLoL = &smem[buf][3][0];

    FragU fah[4], fal[4], fbh[2], fbl[2];
#pragma unroll
    for (int i = 0; i < 4; ++i) {
      const unsigned short* ph = aHiL + (waveRow * 64 + i * 16 + mrow) * 40 + half * 8;
      fah[i].q[0] = *(const U128*)ph;         // K = 8h+0..7    -> VGPR0..3
      fah[i].q[1] = *(const U128*)(ph + 16);  // K = 16+8h+0..7 -> VGPR4..7
      const unsigned short* pl = aLoL + (waveRow * 64 + i * 16 + mrow) * 40 + half * 8;
      fal[i].q[0] = *(const U128*)pl;
      fal[i].q[1] = *(const U128*)(pl + 16);
    }
#pragma unroll
    for (int j = 0; j < 2; ++j) {
      const unsigned short* ph = bHiL + (waveCol * 32 + j * 16 + mrow) * 40 + half * 8;
      fbh[j].q[0] = *(const U128*)ph;         // K = 8h+0..7    -> VGPR0..3
      fbh[j].q[1] = *(const U128*)(ph + 16);  // K = 16+8h+0..7 -> VGPR4..7
      const unsigned short* pl = bLoL + (waveCol * 32 + j * 16 + mrow) * 40 + half * 8;
      fbl[j].q[0] = *(const U128*)pl;
      fbl[j].q[1] = *(const U128*)(pl + 16);
    }

    // bf16x3: acc += lo*hi + hi*lo + hi*hi (smallest first)
#pragma unroll
    for (int i = 0; i < 4; ++i)
#pragma unroll
      for (int j = 0; j < 2; ++j) {
        acc[i][j] = __builtin_amdgcn_wmma_f32_16x16x32_bf16(
            false, fal[i].v, false, fbh[j].v, (short)0, acc[i][j], false, false);
        acc[i][j] = __builtin_amdgcn_wmma_f32_16x16x32_bf16(
            false, fah[i].v, false, fbl[j].v, (short)0, acc[i][j], false, false);
        acc[i][j] = __builtin_amdgcn_wmma_f32_16x16x32_bf16(
            false, fah[i].v, false, fbh[j].v, (short)0, acc[i][j], false, false);
      }
    buf ^= 1;
  }

  // epilogue; C layout: VGPR v -> M = v + 8*half, N = lane&15
#pragma unroll
  for (int j = 0; j < 2; ++j) {
    int col = colBase + waveCol * 32 + j * 16 + mrow;
    float bj = bias[col];
#pragma unroll
    for (int i = 0; i < 4; ++i) {
#pragma unroll
      for (int v = 0; v < 8; ++v) {
        int r = rowBase + waveRow * 64 + i * 16 + v + half * 8;
        float val = acc[i][j][v] + bj;
        if (relu) val = fmaxf(val, 0.f);
        if (outMode == 0) {
          Cf[(size_t)r * N + col] = val;
        } else if (outMode == 1) {
          unsigned short h, l;
          bf16_split(val, h, l);
          cHi[(size_t)r * N + col] = h;
          cLo[(size_t)r * N + col] = l;
        } else {
          // G1 states: chunk-blocked planes inside d_out.
          unsigned short h, l;
          bf16_split(val, h, l);
          int strd = N / Dd;
          int t = r * strd + col / Dd;
          int d = col - (col / Dd) * Dd;
          int c0 = (t / CHc) * CHc;
          size_t idx = (size_t)c0 * 2 * Vv + (size_t)(t - c0) * Dd + d;
          cHi[idx] = h;
          cHi[idx + (size_t)CHc * Dd] = l;
        }
      }
    }
  }
}

// in-place log_softmax over rows of x[rows, V]; one 128-thread block per row
__global__ __launch_bounds__(128)
void logsoftmax_kernel(float* __restrict__ x, int V) {
  float* p = x + (size_t)blockIdx.x * V;
  const int tid = threadIdx.x;
  const int w = tid >> 5, ln = tid & 31;
  __shared__ float redm[4];
  __shared__ float reds[4];

  float m = -INFINITY;
  for (int e = tid; e < V; e += 128) m = fmaxf(m, p[e]);
  for (int off = 16; off > 0; off >>= 1) m = fmaxf(m, __shfl_xor(m, off, 32));
  if (ln == 0) redm[w] = m;
  __syncthreads();
  m = fmaxf(fmaxf(redm[0], redm[1]), fmaxf(redm[2], redm[3]));

  float s = 0.f;
  for (int e = tid; e < V; e += 128) s += expf(p[e] - m);
  for (int off = 16; off > 0; off >>= 1) s += __shfl_xor(s, off, 32);
  if (ln == 0) reds[w] = s;
  __syncthreads();
  s = reds[0] + reds[1] + reds[2] + reds[3];

  float lse = m + logf(s);
  for (int e = tid; e < V; e += 128) p[e] -= lse;
}

__device__ __forceinline__ float lae(float a, float b) {
  float m = fmaxf(a, b);
  float d = fabsf(a - b);
  return m + log1pf(expf(-d));
}

// CTC forward DP: one workgroup per batch; thread s owns alpha[s]; double-buffered
// alpha in LDS; lp(t+1) prefetched while computing step t.
__global__ void ctc_kernel(const float* __restrict__ logprobs, const unsigned char* __restrict__ encm,
                           const int* __restrict__ targets, const unsigned char* __restrict__ tgtm,
                           float* __restrict__ per_loss, float* __restrict__ out_len,
                           int T, int V, int L, int Ssrc, int stride_) {
  const int b = blockIdx.x;
  const int s = threadIdx.x;
  const int S = 2 * L + 1;

  __shared__ float buf0[1025];
  __shared__ float buf1[1025];
  __shared__ int cnt[2];
  if (s < 2) cnt[s] = 0;
  __syncthreads();
  for (int i = s; i < Ssrc; i += blockDim.x) if (encm[b * Ssrc + i]) atomicAdd(&cnt[0], 1);
  for (int i = s; i < L;    i += blockDim.x) if (tgtm[b * L + i])    atomicAdd(&cnt[1], 1);
  __syncthreads();
  const int seqlen = cnt[0];
  const int tl     = cnt[1];
  const int inlen  = stride_ * seqlen;

  int es = 1, skipf = 0;
  if (s < S && (s & 1)) {
    es = targets[b * L + (s >> 1)];
    int p2 = (s >= 3) ? targets[b * L + (s >> 1) - 1] : 1;
    skipf = (es != 1) && (es != p2);
  }

  const size_t base = (size_t)b * T * V;
  if (s < S) {
    float lp0 = logprobs[base + es];
    float a = NEGV;
    if (s == 0) a = lp0;
    else if (s == 1) a = (tl > 0) ? lp0 : NEGV;
    buf0[s] = a;
  }
  __syncthreads();

  float* prev = buf0;
  float* cur  = buf1;
  float lpc = 0.f;
  if (s < S && T > 1) lpc = logprobs[base + (size_t)V + es];

  for (int t = 1; t < T; ++t) {
    float lpn = 0.f;
    if (s < S && (t + 1) < T) lpn = logprobs[base + (size_t)(t + 1) * V + es];  // prefetch
    if (s < S) {
      float a  = prev[s];
      float a1 = (s >= 1) ? prev[s - 1] : NEGV;
      float a2 = (skipf && s >= 2) ? prev[s - 2] : NEGV;
      float nv = lae(lae(a, a1), a2) + lpc;
      cur[s] = (t < inlen) ? nv : a;
    }
    __syncthreads();
    float* tp = prev; prev = cur; cur = tp;
    lpc = lpn;
  }

  if (s == 0) {
    int il = 2 * tl;
    float al = prev[il];
    float ap = (tl > 0) ? prev[il - 1] : NEGV;
    float loss = -lae(al, ap);
    per_loss[b] = loss / fmaxf((float)tl, 1.0f);
    out_len[b] = (float)inlen;
  }
}

__global__ void finalize_kernel(const float* __restrict__ per_loss, float* __restrict__ loss_out, int B) {
  float v = ((int)threadIdx.x < B) ? per_loss[threadIdx.x] : 0.f;
  for (int off = 16; off > 0; off >>= 1) v += __shfl_xor(v, off, 32);
  if (threadIdx.x == 0) {
    float m = v / (float)B;
    if (__builtin_isnan(m) || __builtin_isinf(m)) m = 0.f;
    *loss_out = m;
  }
}

extern "C" void kernel_launch(void* const* d_in, const int* in_sizes, int n_in,
                              void* d_out, int out_size, void* d_ws, size_t ws_size,
                              hipStream_t stream) {
  (void)n_in; (void)out_size;
  const float*         rep     = (const float*)d_in[0];
  const unsigned char* encm    = (const unsigned char*)d_in[1];   // jax bool = 1 byte
  const int*           targets = (const int*)d_in[2];
  const unsigned char* tgtm    = (const unsigned char*)d_in[3];
  const float*         W_exp   = (const float*)d_in[4];
  const float*         b_exp   = (const float*)d_in[5];
  const float*         W1      = (const float*)d_in[6];
  const float*         b1      = (const float*)d_in[7];
  const float*         W2      = (const float*)d_in[8];
  const float*         b2      = (const float*)d_in[9];

  const int B      = 32;
  const int D      = in_sizes[7] / 2;        // b1 = 2D
  const int STRIDE = in_sizes[5] / D;        // b_exp = STRIDE*D
  const int V      = in_sizes[9];            // b2
  const int Ssrc   = in_sizes[1] / B;
  const int L      = in_sizes[2] / B;
  const int H      = 2 * D;
  const int M1     = B * Ssrc;               // 16384
  const int T      = STRIDE * Ssrc;          // 2048
  const int Mtot   = B * T;                  // 65536
  const int NE     = STRIDE * D;             // 2048

  float* out     = (float*)d_out;
  float* outLen  = out + (size_t)Mtot * V;
  float* lossOut = outLen + B;

  // ---- workspace carve-up (ushort units, 256B-aligned slots) ----
  unsigned short* wsp = (unsigned short*)d_ws;
  size_t off = 0;
  auto alloc_us = [&](size_t count) {
    unsigned short* p = wsp + off;
    off += (count + 127) & ~(size_t)127;
    return p;
  };
  unsigned short* wExpHi = alloc_us((size_t)NE * D);
  unsigned short* wExpLo = alloc_us((size_t)NE * D);
  unsigned short* w1Hi   = alloc_us((size_t)H * D);
  unsigned short* w1Lo   = alloc_us((size_t)H * D);
  unsigned short* w2Hi   = alloc_us((size_t)V * H);
  unsigned short* w2Lo   = alloc_us((size_t)V * H);
  unsigned short* repHi  = alloc_us((size_t)M1 * D);
  unsigned short* repLo  = alloc_us((size_t)M1 * D);

  size_t fixedBytes = off * sizeof(unsigned short) + 1024;
  size_t avail = ws_size > fixedBytes ? ws_size - fixedBytes : 0;
  int CH = 8192;                                   // rows per G2/G3 chunk
  while (CH > 512 && (size_t)CH * H * 4 > avail) CH >>= 1;
  unsigned short* hHi = alloc_us((size_t)CH * H);
  unsigned short* hLo = alloc_us((size_t)CH * H);
  float* per_loss = (float*)(wsp + off);

  // ---- one-time operand packing ----
  convert_transpose_kernel<<<(NE * D + 255) / 256, 256, 0, stream>>>(W_exp, wExpHi, wExpLo, D, NE);
  convert_transpose_kernel<<<(H * D + 255) / 256, 256, 0, stream>>>(W1, w1Hi, w1Lo, D, H);
  convert_transpose_kernel<<<(V * H + 255) / 256, 256, 0, stream>>>(W2, w2Hi, w2Lo, H, V);
  convert_plane_kernel<<<(M1 * D + 255) / 256, 256, 0, stream>>>(rep, repHi, repLo, M1 * D);

  // G1: states planes (chunk-blocked) into d_out
  dim3 g1(NE / 128, M1 / 128);
  gemm_wmma_async<<<g1, 256, 0, stream>>>(repHi, repLo, wExpHi, wExpLo, b_exp,
                                          nullptr, (unsigned short*)out, nullptr,
                                          NE, D, 0, 2, D, V, CH);

  // chunked G2/G3; h planes stay L2-resident; G3 chunk c overwrites exactly the
  // states plane area chunk c already consumed.
  for (int c0 = 0; c0 < Mtot; c0 += CH) {
    const unsigned short* aHi = (const unsigned short*)out + (size_t)c0 * 2 * V;
    const unsigned short* aLo = aHi + (size_t)CH * D;
    dim3 g2(H / 128, CH / 128);
    gemm_wmma_async<<<g2, 256, 0, stream>>>(aHi, aLo, w1Hi, w1Lo, b1,
                                            nullptr, hHi, hLo,
                                            H, D, 1, 1, D, V, CH);
    dim3 g3(V / 128, CH / 128);
    gemm_wmma_async<<<g3, 256, 0, stream>>>(hHi, hLo, w2Hi, w2Lo, b2,
                                            out + (size_t)c0 * V, nullptr, nullptr,
                                            V, H, 0, 0, D, V, CH);
  }

  // logits -> logprobs in place
  logsoftmax_kernel<<<Mtot, 128, 0, stream>>>(out, V);

  // CTC DP
  int S = 2 * L + 1;
  int ctcThreads = ((S + 31) / 32) * 32;
  if (ctcThreads > 1024) ctcThreads = 1024;
  ctc_kernel<<<B, ctcThreads, 0, stream>>>(out, encm, targets, tgtm, per_loss, outLen,
                                           T, V, L, Ssrc, STRIDE);
  finalize_kernel<<<1, 32, 0, stream>>>(per_loss, lossOut, B);
}
